// T5Encoder_79302276153519
// MI455X (gfx1250) — matrix operand order repaired and checked by reading, not compiled, not run on hardware
//
#include <hip/hip_runtime.h>
#include <hip/hip_bf16.h>

// ---------------- problem constants ----------------
constexpr int Bsz    = 4;
constexpr int Nseq   = 1024;
constexpr int Dm     = 768;
constexpr int Hh     = 12;
constexpr int DHd    = 64;
constexpr int DEPTHn = 6;
constexpr int FFd    = 3072;
constexpr int INNERd = Hh * DHd;        // 768
constexpr int Mrows  = Bsz * Nseq;      // 4096

// ---------------- vector types ----------------
typedef __attribute__((ext_vector_type(16))) __bf16         v16bf;
typedef __attribute__((ext_vector_type(8)))  float          v8f;
typedef __attribute__((ext_vector_type(8)))  unsigned short us8;
typedef __attribute__((ext_vector_type(4)))  unsigned short us4;
typedef __attribute__((ext_vector_type(4)))  float          f4;
typedef __attribute__((ext_vector_type(4)))  unsigned int   u32x4;
typedef __attribute__((ext_vector_type(8)))  int            i32x8;
typedef __attribute__((ext_vector_type(4)))  int            i32x4;

union AccU  { v8f v;   float f[8]; };
union FragU { v16bf v; us8 h[2]; unsigned short u[16]; };

__device__ __forceinline__ unsigned short f2bf(float f) {
    __bf16 b = (__bf16)f;                      // hardware cvt
    return __builtin_bit_cast(unsigned short, b);
}

// =====================================================================
// TDM: load a 128-row x 32-col bf16 tile (row stride = strideElems) from
// global into LDS with a 16B pad every row (pitch 40 ushort = 80B).
//   pad_interval = 3 (16 DWORDs = 64B = one 32-elem bf16 row)
//   pad_amount   = 3 (4 DWORDs = 16B)
// =====================================================================
__device__ __forceinline__ void tdm_load_tile(unsigned ldsAddr,
                                              const unsigned short* g,
                                              int strideElems)
{
    unsigned long long ga = (unsigned long long)(const void*)g;
    u32x4 g0;
    g0[0] = 1u;                                        // count=1, user descriptor
    g0[1] = ldsAddr;                                   // LDS byte address
    g0[2] = (unsigned)ga;                              // global_addr[31:0]
    g0[3] = (unsigned)(ga >> 32) | 0x80000000u;        // addr[56:32] | type=2
    i32x8 g1;
    g1[0] = (1 << 16) | (1 << 20) | (3 << 22) | (3 << 25); // dsz=2B,pad_en,int=16DW,amt=4DW
    g1[1] = (32 << 16);                                // tensor_dim0[15:0] = 32
    g1[2] = (128 << 16);                               // dim0 hi=0 | tensor_dim1[15:0]=128
    g1[3] = (32 << 16);                                // dim1 hi=0 | tile_dim0 = 32
    g1[4] = 128;                                       // tile_dim1 = 128, tile_dim2 = 0
    g1[5] = strideElems;                               // tensor_dim0_stride[31:0]
    g1[6] = 0;                                         // stride hi, dim1_stride lo
    g1[7] = 0;
    i32x4 g2 = (i32x4)0;
    i32x4 g3 = (i32x4)0;
    i32x8 g4 = (i32x8)0;                               // unused trailing group (6-arg form)
    __builtin_amdgcn_tensor_load_to_lds(g0, g1, g2, g3, g4, 0);
}

// =====================================================================
// GEMM: C[M,N] = op( A[M,K] @ W[N,K]^T ), A/W bf16, TDM-staged, WMMA bf16
// flags: 1 = +R residual(f32), 2 = relu, 4 = write bf16 copy Cb, 8 = skip f32 C
// block = 256 threads (8 waves), tile 128x128, K-step 32, double-buffered LDS
// =====================================================================
__global__ __launch_bounds__(256)
void gemm_bf16_kernel(const unsigned short* __restrict__ A,
                      const unsigned short* __restrict__ W,
                      const float* __restrict__ R,
                      float* __restrict__ C, unsigned short* __restrict__ Cb,
                      int M, int Nn, int K, int flags)
{
    __shared__ unsigned short As[2][128][40];    // 20,480 B
    __shared__ unsigned short Bs[2][128][40];    // 20,480 B

    const int t    = threadIdx.x;
    const int wave = t >> 5, lane = t & 31, lo = lane & 15, hi = lane >> 4;
    const int m0   = blockIdx.y * 128, n0 = blockIdx.x * 128;
    const int wm   = (wave >> 2) * 64, wn = (wave & 3) * 32;

    AccU acc[4][2];
#pragma unroll
    for (int i = 0; i < 4; ++i)
#pragma unroll
        for (int j = 0; j < 2; ++j)
#pragma unroll
            for (int e = 0; e < 8; ++e) acc[i][j].f[e] = 0.f;

    // TDM issue helper (wave 0 only): stage tile for K-offset k into buffer (k/32)&1
    auto issue = [&](int k) {
        int buf = (k >> 5) & 1;
        tdm_load_tile((unsigned)(unsigned long long)(const void*)&As[buf][0][0],
                      A + (size_t)m0 * K + k, K);
        tdm_load_tile((unsigned)(unsigned long long)(const void*)&Bs[buf][0][0],
                      W + (size_t)n0 * K + k, K);
    };

    if (wave == 0) issue(0);

    for (int k0 = 0; k0 < K; k0 += 32) {
        if (wave == 0) __builtin_amdgcn_s_wait_tensorcnt(0);   // current tile landed
        __syncthreads();
        if (wave == 0 && k0 + 32 < K) issue(k0 + 32);          // overlap next DMA
        const int buf = (k0 >> 5) & 1;

        // A fragments: lane lo = row M; K elems {hi*8..+7, 16+hi*8..+7}
        FragU af[4], bfr[2];
#pragma unroll
        for (int i = 0; i < 4; ++i) {
            const unsigned short* p = &As[buf][wm + i * 16 + lo][0];
            af[i].h[0] = *(const us8*)(p + hi * 8);
            af[i].h[1] = *(const us8*)(p + 16 + hi * 8);
        }
        // B fragments: lane lo = column N; K elems {hi*8..+7, 16+hi*8..+7}
#pragma unroll
        for (int j = 0; j < 2; ++j) {
            const unsigned short* p = &Bs[buf][wn + j * 16 + lo][0];
            bfr[j].h[0] = *(const us8*)(p + hi * 8);
            bfr[j].h[1] = *(const us8*)(p + 16 + hi * 8);
        }
#pragma unroll
        for (int i = 0; i < 4; ++i)
#pragma unroll
            for (int j = 0; j < 2; ++j)
                acc[i][j].v = __builtin_amdgcn_wmma_f32_16x16x32_bf16(
                    false, af[i].v, false, bfr[j].v, (short)0, acc[i][j].v, false, false);
        __syncthreads();   // all reads of buf done before its next TDM overwrite
    }

    // epilogue: C/D layout — VGPR r : lanes0-15 -> M=r, lanes16-31 -> M=8+r, N=lane%16
#pragma unroll
    for (int i = 0; i < 4; ++i)
#pragma unroll
        for (int j = 0; j < 2; ++j)
#pragma unroll
            for (int r = 0; r < 8; ++r) {
                int row = m0 + wm + i * 16 + r + hi * 8;
                int col = n0 + wn + j * 16 + lo;
                float val = acc[i][j].f[r];
                if (flags & 2) val = fmaxf(val, 0.f);
                if (flags & 1) val += R[(size_t)row * Nn + col];
                if (!(flags & 8)) C[(size_t)row * Nn + col] = val;
                if (flags & 4)  Cb[(size_t)row * Nn + col] = f2bf(val);
            }
}

// =====================================================================
// Flash attention: block = 4 waves = 64 q-rows of one (b,h)
// q,k,v,out are bf16 [B*N, INNER] row-major (col = h*64+d)
// =====================================================================
__global__ __launch_bounds__(128)
void attn_kernel(const unsigned short* __restrict__ q,
                 const unsigned short* __restrict__ k,
                 const unsigned short* __restrict__ v,
                 const unsigned char* __restrict__ mask,
                 unsigned short* __restrict__ out, float scale)
{
    __shared__ unsigned short Kl[32][72];      // [j][d] row-major
    __shared__ unsigned short Vt[64][40];      // [d][j] transposed
    __shared__ unsigned short Pl[4][16][40];   // per-wave P transpose scratch

    const int t = threadIdx.x, wave = t >> 5, lane = t & 31, lo = lane & 15, hi = lane >> 4;
    const int q0 = blockIdx.x * 64, hh = blockIdx.y, b = blockIdx.z;
    const int colbase = hh * DHd;
    const int qrow = b * Nseq + q0 + wave * 16 + lo;

    // resident q A-fragments: direct global b128 loads (bf16 already laid out)
    FragU qa[2];
#pragma unroll
    for (int f = 0; f < 2; ++f) {
        const unsigned short* p = q + (size_t)qrow * INNERd + colbase + f * 32 + hi * 8;
        qa[f].h[0] = *(const us8*)p;
        qa[f].h[1] = *(const us8*)(p + 16);
    }

    float mrow[8], lrow[8];
    AccU Ot[4];
#pragma unroll
    for (int r = 0; r < 8; ++r) { mrow[r] = -3.0e38f; lrow[r] = 0.f; }
#pragma unroll
    for (int tt = 0; tt < 4; ++tt)
#pragma unroll
        for (int e = 0; e < 8; ++e) Ot[tt].f[e] = 0.f;

    for (int j0 = 0; j0 < Nseq; j0 += 32) {
        __syncthreads();
        // stage K (row major copy) and V (transposed), bf16->bf16
#pragma unroll
        for (int i = 0; i < 2; ++i) {
            int id = t + i * 128;
            int jr = id >> 3;            // 0..31
            int c8 = (id & 7) * 8;       // 0..56
            const size_t g = (size_t)(b * Nseq + j0 + jr) * INNERd + colbase + c8;
            us8 kv = *(const us8*)(k + g);
            us8 vv = *(const us8*)(v + g);
            *(us8*)&Kl[jr][c8] = kv;
#pragma unroll
            for (int jj = 0; jj < 8; ++jj) Vt[c8 + jj][jr] = vv[jj];
        }
        __syncthreads();

        // sim tiles for j-subtiles 0 and 16 (K-dim = d, 2 WMMAs each)
        AccU sub[2];
#pragma unroll
        for (int s2 = 0; s2 < 2; ++s2) {
            FragU kb0, kb1;
            const unsigned short* p0 = &Kl[s2 * 16 + lo][0];
            kb0.h[0] = *(const us8*)(p0 + hi * 8);
            kb0.h[1] = *(const us8*)(p0 + 16 + hi * 8);
            kb1.h[0] = *(const us8*)(p0 + 32 + hi * 8);
            kb1.h[1] = *(const us8*)(p0 + 48 + hi * 8);
            AccU sacc;
#pragma unroll
            for (int e = 0; e < 8; ++e) sacc.f[e] = 0.f;
            sacc.v = __builtin_amdgcn_wmma_f32_16x16x32_bf16(false, qa[0].v, false, kb0.v,
                                                             (short)0, sacc.v, false, false);
            sacc.v = __builtin_amdgcn_wmma_f32_16x16x32_bf16(false, qa[1].v, false, kb1.v,
                                                             (short)0, sacc.v, false, false);
            int colg = j0 + s2 * 16 + lo;
            bool mk = mask[b * Nseq + colg] != 0;
#pragma unroll
            for (int r = 0; r < 8; ++r) {
                float val = sacc.f[r] * scale;
                sub[s2].f[r] = mk ? val : -3.0e38f;
            }
        }

        // online softmax (row r+8*hi lives in a 16-lane group)
        float mx[8], corr[8], rs[8];
#pragma unroll
        for (int r = 0; r < 8; ++r) mx[r] = fmaxf(sub[0].f[r], sub[1].f[r]);
        for (int off = 1; off < 16; off <<= 1)
#pragma unroll
            for (int r = 0; r < 8; ++r) mx[r] = fmaxf(mx[r], __shfl_xor(mx[r], off, 32));
#pragma unroll
        for (int r = 0; r < 8; ++r) {
            float mn = fmaxf(mrow[r], mx[r]);
            corr[r] = __expf(mrow[r] - mn);
            mrow[r] = mn;
        }
#pragma unroll
        for (int s2 = 0; s2 < 2; ++s2)
#pragma unroll
            for (int r = 0; r < 8; ++r) sub[s2].f[r] = __expf(sub[s2].f[r] - mrow[r]);
#pragma unroll
        for (int r = 0; r < 8; ++r) rs[r] = sub[0].f[r] + sub[1].f[r];
        for (int off = 1; off < 16; off <<= 1)
#pragma unroll
            for (int r = 0; r < 8; ++r) rs[r] += __shfl_xor(rs[r], off, 32);
#pragma unroll
        for (int r = 0; r < 8; ++r) lrow[r] = lrow[r] * corr[r] + rs[r];
#pragma unroll
        for (int tt = 0; tt < 4; ++tt)
#pragma unroll
            for (int r = 0; r < 8; ++r) Ot[tt].f[r] *= corr[r];

        // P (C-layout) -> A-fragment layout via per-wave LDS scratch
#pragma unroll
        for (int s2 = 0; s2 < 2; ++s2)
#pragma unroll
            for (int r = 0; r < 8; ++r)
                Pl[wave][r + hi * 8][s2 * 16 + lo] = f2bf(sub[s2].f[r]);
        asm volatile("s_wait_dscnt 0" ::: "memory");
        FragU pa;
        const unsigned short* pp = &Pl[wave][lo][0];
        pa.h[0] = *(const us8*)(pp + hi * 8);
        pa.h[1] = *(const us8*)(pp + 16 + hi * 8);

        // O[16x64] += P[16x32] @ V[32x64]
#pragma unroll
        for (int tt = 0; tt < 4; ++tt) {
            FragU vb;
            const unsigned short* vp = &Vt[tt * 16 + lo][0];
            vb.h[0] = *(const us8*)(vp + hi * 8);
            vb.h[1] = *(const us8*)(vp + 16 + hi * 8);
            Ot[tt].v = __builtin_amdgcn_wmma_f32_16x16x32_bf16(false, pa.v, false, vb.v,
                                                               (short)0, Ot[tt].v, false, false);
        }
    }

    // epilogue: normalize, write bf16 [B*N, INNER]
#pragma unroll
    for (int tt = 0; tt < 4; ++tt)
#pragma unroll
        for (int r = 0; r < 8; ++r) {
            int row = b * Nseq + q0 + wave * 16 + r + hi * 8;
            int col = colbase + tt * 16 + lo;
            out[(size_t)row * INNERd + col] = f2bf(Ot[tt].f[r] / lrow[r]);
        }
}

// =====================================================================
// f32 -> bf16 bulk convert (n multiple of 1024; 4 elems/thread)
// =====================================================================
__global__ __launch_bounds__(256)
void cvt_bf16_kernel(const float* __restrict__ src, unsigned short* __restrict__ dst)
{
    size_t i = ((size_t)blockIdx.x * 256 + threadIdx.x) * 4;
    f4 v = *(const f4*)(src + i);
    us4 o;
#pragma unroll
    for (int j = 0; j < 4; ++j) o[j] = f2bf(v[j]);
    *(us4*)(dst + i) = o;
}

// =====================================================================
// embedding gather: one block per row; writes f32 + bf16 copies
// =====================================================================
__global__ __launch_bounds__(256)
void embed_kernel(const int* __restrict__ tokens, const float* __restrict__ embed,
                  float* __restrict__ x, unsigned short* __restrict__ xb)
{
    int row = blockIdx.x;
    int tok = tokens[row];
#pragma unroll
    for (int i = 0; i < 3; ++i) {
        int c = threadIdx.x + i * 256;
        float v = embed[(size_t)tok * Dm + c];
        x[(size_t)row * Dm + c]  = v;
        xb[(size_t)row * Dm + c] = f2bf(v);
    }
}

// =====================================================================
// T5 RMSNorm: one block per row (f32 in, f32 out)
// =====================================================================
__global__ __launch_bounds__(256)
void rmsnorm_kernel(const float* __restrict__ x, const float* __restrict__ w,
                    float* __restrict__ out)
{
    __shared__ float red[8];
    int row = blockIdx.x, t = threadIdx.x;
    float vals[3];
    float ss = 0.f;
#pragma unroll
    for (int i = 0; i < 3; ++i) {
        int c = t + i * 256;
        vals[i] = x[(size_t)row * Dm + c];
        ss += vals[i] * vals[i];
    }
    for (int off = 16; off > 0; off >>= 1) ss += __shfl_xor(ss, off, 32);
    if ((t & 31) == 0) red[t >> 5] = ss;
    __syncthreads();
    float tot = 0.f;
#pragma unroll
    for (int i = 0; i < 8; ++i) tot += red[i];
    float rs = rsqrtf(tot * (1.0f / Dm) + 1e-6f);
#pragma unroll
    for (int i = 0; i < 3; ++i) {
        int c = t + i * 256;
        out[(size_t)row * Dm + c] = w[c] * vals[i] * rs;
    }
}

// =====================================================================
// host driver
// =====================================================================
extern "C" void kernel_launch(void* const* d_in, const int* in_sizes, int n_in,
                              void* d_out, int out_size, void* d_ws, size_t ws_size,
                              hipStream_t stream)
{
    (void)in_sizes; (void)n_in; (void)out_size; (void)ws_size;

    const int*           tokens = (const int*)d_in[0];
    const unsigned char* mask   = (const unsigned char*)d_in[1];
    const float*         embedw = (const float*)d_in[2];
    const float*         wq     = (const float*)d_in[3];
    const float*         wk     = (const float*)d_in[4];
    const float*         wv     = (const float*)d_in[5];
    const float*         wo     = (const float*)d_in[6];
    const float*         wi     = (const float*)d_in[7];
    const float*         wof    = (const float*)d_in[8];
    const float*         lnw    = (const float*)d_in[9];
    float*               outp   = (float*)d_out;

    // workspace layout
    const size_t nXD  = (size_t)Mrows * Dm;      // 4096*768
    const size_t nXF  = (size_t)Mrows * FFd;     // 4096*3072
    const size_t nW   = (size_t)INNERd * Dm;     // 589824 (per qkvo matrix)
    const size_t nWF  = (size_t)FFd * Dm;        // 2359296 (per ff matrix)

    char* p = (char*)d_ws;
    float* xA           = (float*)p;           p += nXD * 4;
    float* xB           = (float*)p;           p += nXD * 4;
    unsigned short* xAb = (unsigned short*)p;  p += nXD * 2;
    unsigned short* xBb = (unsigned short*)p;  p += nXD * 2;
    unsigned short* qb  = (unsigned short*)p;  p += nXD * 2;
    unsigned short* kb  = (unsigned short*)p;  p += nXD * 2;
    unsigned short* vb  = (unsigned short*)p;  p += nXD * 2;
    unsigned short* ob  = (unsigned short*)p;  p += nXD * 2;
    unsigned short* hb  = (unsigned short*)p;  p += nXF * 2;
    unsigned short* wqb = (unsigned short*)p;  p += nW * 2;
    unsigned short* wkb = (unsigned short*)p;  p += nW * 2;
    unsigned short* wvb = (unsigned short*)p;  p += nW * 2;
    unsigned short* wob = (unsigned short*)p;  p += nW * 2;
    unsigned short* wib = (unsigned short*)p;  p += nWF * 2;
    unsigned short* wfb = (unsigned short*)p;  /* end */

    const float scale = 0.125f;                  // DH^-0.5
    dim3 blk256(256), blk128(128);
    dim3 gProj(INNERd / 128, Mrows / 128);       // 6 x 32
    dim3 gD(Dm / 128, Mrows / 128);              // 6 x 32
    dim3 gFF(FFd / 128, Mrows / 128);            // 24 x 32
    dim3 gAttn(Nseq / 64, Hh, Bsz);              // 16 x 12 x 4

    embed_kernel<<<Mrows, blk256, 0, stream>>>(tokens, embedw, xA, xAb);

    for (int l = 0; l < DEPTHn; ++l) {
        // per-layer weight conversion f32 -> bf16 (reused buffer)
        cvt_bf16_kernel<<<nW  / 1024, blk256, 0, stream>>>(wq  + (size_t)l * nW,  wqb);
        cvt_bf16_kernel<<<nW  / 1024, blk256, 0, stream>>>(wk  + (size_t)l * nW,  wkb);
        cvt_bf16_kernel<<<nW  / 1024, blk256, 0, stream>>>(wv  + (size_t)l * nW,  wvb);
        cvt_bf16_kernel<<<nW  / 1024, blk256, 0, stream>>>(wo  + (size_t)l * nW,  wob);
        cvt_bf16_kernel<<<nWF / 1024, blk256, 0, stream>>>(wi  + (size_t)l * nWF, wib);
        cvt_bf16_kernel<<<nWF / 1024, blk256, 0, stream>>>(wof + (size_t)l * nWF, wfb);

        // q,k,v (bf16 only)
        gemm_bf16_kernel<<<gProj, blk256, 0, stream>>>(xAb, wqb, nullptr, nullptr, qb,
                                                       Mrows, INNERd, Dm, 4 | 8);
        gemm_bf16_kernel<<<gProj, blk256, 0, stream>>>(xAb, wkb, nullptr, nullptr, kb,
                                                       Mrows, INNERd, Dm, 4 | 8);
        gemm_bf16_kernel<<<gProj, blk256, 0, stream>>>(xAb, wvb, nullptr, nullptr, vb,
                                                       Mrows, INNERd, Dm, 4 | 8);

        attn_kernel<<<gAttn, blk128, 0, stream>>>(qb, kb, vb, mask, ob, scale);

        // x = x + o @ wo^T     (f32 xB + bf16 xBb)
        gemm_bf16_kernel<<<gD, blk256, 0, stream>>>(ob, wob, xA, xB, xBb,
                                                    Mrows, Dm, INNERd, 1 | 4);
        // h = relu(x @ wi^T)   (bf16 only)
        gemm_bf16_kernel<<<gFF, blk256, 0, stream>>>(xBb, wib, nullptr, nullptr, hb,
                                                     Mrows, FFd, Dm, 2 | 4 | 8);
        // x = x + h @ wof^T    (f32 xA + bf16 xAb)
        gemm_bf16_kernel<<<gD, blk256, 0, stream>>>(hb, wfb, xB, xA, xAb,
                                                    Mrows, Dm, FFd, 1 | 4);
    }

    rmsnorm_kernel<<<Mrows, blk256, 0, stream>>>(xA, lnw, outp);
}
